// LSTM_45011257262795
// MI455X (gfx1250) — matrix operand order repaired and checked by reading, not compiled, not run on hardware
//
#include <hip/hip_runtime.h>
#include <hip/hip_bf16.h>
#include <math.h>

// LSTM  B=32, T=1024, D=H=512  — CDNA5 (gfx1250) wave32 WMMA bf16 implementation.
//
//   prep_kernel       : W_ih,W_hh fp32 -> bf16 (row-major [2048][512]); bias = b_ih+b_hh;
//                       zero h double-buffers, running c (d_out c_n slot), grid-barrier counter.
//   xg_gemm_kernel    : xg[B*T,2048] = x @ W_ih^T + bias via v_wmma_f32_16x16x32_bf16
//                       (wave = 16-row M tile x four 16-col N tiles, K=512).
//   lstm_scan_kernel  : ONE persistent kernel runs all T=1024 steps. 64 blocks x 256 threads
//                       (8 waves = 4 gates x 2 K-halves). Each wave pins its 8 W_hh B-fragments
//                       in VGPRs for the whole scan (64 VGPRs of weights -> zero weight traffic
//                       in the serial loop), software-pipelines the h A-fragment loads, does
//                       8 WMMAs, reduces K-half partials + fuses sigmoid/tanh through LDS, and
//                       synchronizes the grid with a global atomic counter + s_sleep spin
//                       (no per-step launch overhead).
//
// Workspace (~260.5 MB): W_ih bf16 (2MB) | W_hh bf16 (2MB) | bias f32 (8KB) |
//                        h0,h1 bf16 (32KB each) | counter | 256B-aligned xg f32 (256MB).

typedef __bf16 bf16;
typedef __attribute__((ext_vector_type(16))) __bf16 v16bf;
typedef __attribute__((ext_vector_type(8)))  float  v8f;

#define B_ 32
#define T_ 1024
#define D_ 512
#define H_ 512
#define G_ 2048   // 4*H
#define NBLK_ 64  // persistent scan blocks (2 M-tiles x 32 hidden-col tiles)

union BF16Frag { uint4 q[2]; v16bf v; };

// B fragment of B[k][n] = W[n][k] for a 32x16 (KxN) tile at (kt, n0).
// Lane l: n = n0 + (l&15); elems j<8 -> K = kt+kh+j, j>=8 -> K = kt+16+kh+(j-8), kh=(l>>4)*8.
__device__ inline v16bf load_b_frag(const bf16* __restrict__ W, int n0, int kt, int lane) {
  int n = n0 + (lane & 15);
  int koff = (lane >> 4) * 8;
  const bf16* p = W + (size_t)n * 512 + kt + koff;
  BF16Frag f;
  f.q[0] = *(const uint4*)(p);
  f.q[1] = *(const uint4*)(p + 16);
  return f.v;
}

// A fragment (16x32 bf16, row-major [rows][512]).
// Lane l: m = m0 + (l&15); elems j<8 -> K = kt+kh+j, j>=8 -> K = kt+16+kh+(j-8), kh=(l>>4)*8.
__device__ inline v16bf load_a_frag_bf16(const bf16* __restrict__ Hm, int m0, int kt, int lane) {
  int m = m0 + (lane & 15);
  int kh = (lane >> 4) * 8;
  const bf16* p = Hm + (size_t)m * 512 + kt + kh;
  BF16Frag f;
  f.q[0] = *(const uint4*)(p);
  f.q[1] = *(const uint4*)(p + 16);
  return f.v;
}

// A fragment with on-the-fly fp32 -> bf16 convert (for x).
__device__ inline v16bf load_a_frag_f32(const float* __restrict__ X, int m0, int kt, int lane) {
  int m = m0 + (lane & 15);
  int kh = (lane >> 4) * 8;
  const float* p = X + (size_t)m * 512 + kt + kh;
  float4 f0 = *(const float4*)(p);
  float4 f1 = *(const float4*)(p + 4);
  float4 f2 = *(const float4*)(p + 16);
  float4 f3 = *(const float4*)(p + 20);
  v16bf a;
  a[0]=(bf16)f0.x;  a[1]=(bf16)f0.y;  a[2]=(bf16)f0.z;  a[3]=(bf16)f0.w;
  a[4]=(bf16)f1.x;  a[5]=(bf16)f1.y;  a[6]=(bf16)f1.z;  a[7]=(bf16)f1.w;
  a[8]=(bf16)f2.x;  a[9]=(bf16)f2.y;  a[10]=(bf16)f2.z; a[11]=(bf16)f2.w;
  a[12]=(bf16)f3.x; a[13]=(bf16)f3.y; a[14]=(bf16)f3.z; a[15]=(bf16)f3.w;
  return a;
}

__global__ __launch_bounds__(256) void prep_kernel(
    const float* __restrict__ Wih, const float* __restrict__ Whh,
    const float* __restrict__ bih, const float* __restrict__ bhh,
    bf16* __restrict__ Wih_b, bf16* __restrict__ Whh_b,
    float* __restrict__ bias, bf16* __restrict__ h0, bf16* __restrict__ h1,
    float* __restrict__ cbuf, unsigned* __restrict__ counter) {
  int i = blockIdx.x * 256 + threadIdx.x;
  if (i < G_ * D_) { Wih_b[i] = (bf16)Wih[i]; Whh_b[i] = (bf16)Whh[i]; }
  if (i < G_)      { bias[i] = bih[i] + bhh[i]; }
  if (i < B_ * H_) { h0[i] = (bf16)0.0f; h1[i] = (bf16)0.0f; cbuf[i] = 0.0f; }
  if (i == 0)      { *counter = 0u; }
}

// xg = x @ W_ih^T + bias.  M=32768, N=2048, K=512.  8192 blocks x 256 threads (8 waves).
__global__ __launch_bounds__(256) void xg_gemm_kernel(
    const float* __restrict__ X,     // [B*T, 512]
    const bf16*  __restrict__ Wih,   // [2048, 512] bf16
    const float* __restrict__ bias,  // [2048]
    float* __restrict__ xg)          // [B*T, 2048]
{
  int lane = threadIdx.x & 31;
  int warp = threadIdx.x >> 5;
  int w = blockIdx.x * 8 + warp;
  int m0 = (w >> 5) * 16;
  int n0 = (w & 31) * 64;

  v8f acc[4] = {};
  for (int kt = 0; kt < 512; kt += 32) {
    v16bf a = load_a_frag_f32(X, m0, kt, lane);       // A reused across 4 N-subtiles
#pragma unroll
    for (int s = 0; s < 4; ++s) {
      v16bf b = load_b_frag(Wih, n0 + s * 16, kt, lane);
      acc[s] = __builtin_amdgcn_wmma_f32_16x16x32_bf16(
                   false, a, false, b, (short)0, acc[s], false, false);
    }
  }
  int nl = lane & 15;
  int mbase = m0 + ((lane >> 4) ? 8 : 0);
#pragma unroll
  for (int s = 0; s < 4; ++s) {
    int n = n0 + s * 16 + nl;
    float bs = bias[n];
#pragma unroll
    for (int r = 0; r < 8; ++r)
      xg[(size_t)(mbase + r) * G_ + n] = acc[s][r] + bs;
  }
}

// Persistent scan: all 1024 steps in one launch.
// Block: mtile = blk>>5 (batch rows), n0 = (blk&31)*16 (hidden cols).
// Wave w (0..7): gate = w>>1 (i,f,g,o), K-half = (w&1)*256. W_hh fragments live in VGPRs.
__global__ __launch_bounds__(256) void lstm_scan_kernel(
    const float* __restrict__ xg,    // [B*T, 2048] (bias already included)
    const bf16*  __restrict__ Whh,   // [2048, 512] bf16
    bf16* __restrict__ h0,           // [32, 512] bf16 double buffer
    bf16* __restrict__ h1,           // [32, 512] bf16 double buffer
    float* __restrict__ cbuf,        // [32, 512] f32 running c (== d_out c_n slot)
    float* __restrict__ out,         // [B, T, H]
    float* __restrict__ hn,          // [32, 512] (d_out h_n slot)
    unsigned* __restrict__ counter)  // grid barrier (prep zeroes it)
{
  __shared__ float part[8][16][16];  // per-wave partial tiles: [gate*2 + khalf][m][n]
  int lane  = threadIdx.x & 31;
  int warp  = threadIdx.x >> 5;      // 0..7
  int gate  = warp >> 1;             // 0..3 (i,f,g,o)
  int kbase = (warp & 1) * 256;      // K-half
  int m0    = (blockIdx.x >> 5) * 16;
  int n0    = (blockIdx.x & 31) * 16;

  // Pin this wave's W_hh slice in registers for the whole scan (8 x v16bf = 64 VGPRs).
  v16bf wb[8];
#pragma unroll
  for (int kk = 0; kk < 8; ++kk)
    wb[kk] = load_b_frag(Whh, gate * 512 + n0, kbase + kk * 32, lane);

  int nl = lane & 15;
  int rbase = (lane >> 4) ? 8 : 0;
  int m_e = threadIdx.x >> 4;        // elementwise cell: one per thread
  int n_e = threadIdx.x & 15;
  int bi = m0 + m_e;
  int hcol = n0 + n_e;
  int ci = bi * 512 + hcol;
  const float* xrow = xg + (size_t)bi * T_ * G_;
  float* orow = out + (size_t)bi * T_ * H_;

  for (int t = 0; t < T_; ++t) {
    const bf16* hp = (t & 1) ? h1 : h0;
    bf16*       hx = (t & 1) ? h0 : h1;

    // 8 WMMAs over this wave's K-half; depth-2 pipeline on the h A-fragments.
    v8f acc = {};
    v16bf a = load_a_frag_bf16(hp, m0, kbase, lane);
#pragma unroll
    for (int kk = 0; kk < 8; ++kk) {
      v16bf an;
      if (kk < 7) an = load_a_frag_bf16(hp, m0, kbase + (kk + 1) * 32, lane);
      acc = __builtin_amdgcn_wmma_f32_16x16x32_bf16(
                false, a, false, wb[kk], (short)0, acc, false, false);
      if (kk < 7) a = an;
    }
#pragma unroll
    for (int r = 0; r < 8; ++r) part[warp][rbase + r][nl] = acc[r];
    __syncthreads();

    // Fused gate math: 256 threads, one (m,n) cell each.
    {
      const float* xr = xrow + (size_t)t * G_;
      float pi = part[0][m_e][n_e] + part[1][m_e][n_e] + xr[hcol];
      float pf = part[2][m_e][n_e] + part[3][m_e][n_e] + xr[512 + hcol];
      float pg = part[4][m_e][n_e] + part[5][m_e][n_e] + xr[1024 + hcol];
      float po = part[6][m_e][n_e] + part[7][m_e][n_e] + xr[1536 + hcol];
      float iv = 1.0f / (1.0f + __expf(-pi));
      float fv = 1.0f / (1.0f + __expf(-pf));
      float gv = tanhf(pg);
      float ov = 1.0f / (1.0f + __expf(-po));
      float cn = fv * cbuf[ci] + iv * gv;
      float hv = ov * tanhf(cn);
      cbuf[ci] = cn;
      hx[ci] = (bf16)hv;
      orow[(size_t)t * H_ + hcol] = hv;
      if (t == T_ - 1) hn[ci] = hv;
    }

    // Grid-wide barrier: all h writes of step t visible before any block starts t+1.
    __syncthreads();
    if (threadIdx.x == 0) {
      __threadfence();
      atomicAdd(counter, 1u);
      unsigned target = (unsigned)NBLK_ * (unsigned)(t + 1);
      while (atomicAdd(counter, 0u) < target) { __builtin_amdgcn_s_sleep(2); }
      __threadfence();
    }
    __syncthreads();
  }
}

extern "C" void kernel_launch(void* const* d_in, const int* in_sizes, int n_in,
                              void* d_out, int out_size, void* d_ws, size_t ws_size,
                              hipStream_t stream) {
  const float* x   = (const float*)d_in[0];  // [32,1024,512]
  const float* Wih = (const float*)d_in[1];  // [2048,512]
  const float* Whh = (const float*)d_in[2];  // [2048,512]
  const float* bih = (const float*)d_in[3];  // [2048]
  const float* bhh = (const float*)d_in[4];  // [2048]
  (void)in_sizes; (void)n_in; (void)out_size; (void)ws_size;

  char* ws = (char*)d_ws;
  bf16* Wih_b = (bf16*)ws;        ws += (size_t)G_ * D_ * sizeof(bf16);   // 2 MB
  bf16* Whh_b = (bf16*)ws;        ws += (size_t)G_ * H_ * sizeof(bf16);   // 2 MB
  float* bias = (float*)ws;       ws += (size_t)G_ * sizeof(float);       // 8 KB
  bf16* h0    = (bf16*)ws;        ws += (size_t)B_ * H_ * sizeof(bf16);   // 32 KB
  bf16* h1    = (bf16*)ws;        ws += (size_t)B_ * H_ * sizeof(bf16);   // 32 KB
  unsigned* counter = (unsigned*)ws; ws += 256;                            // barrier counter
  ws = (char*)(((uintptr_t)ws + 255) & ~(uintptr_t)255);
  float* xg   = (float*)ws;                                                // 256 MB

  float* out = (float*)d_out;                  // [B,T,H]
  float* hn  = out + (size_t)B_ * T_ * H_;     // [1,B,H]
  float* cn  = hn + (size_t)B_ * H_;           // [1,B,H] — doubles as running c

  prep_kernel<<<(G_ * D_ + 255) / 256, 256, 0, stream>>>(
      Wih, Whh, bih, bhh, Wih_b, Whh_b, bias, h0, h1, cn, counter);

  xg_gemm_kernel<<<8192, 256, 0, stream>>>(x, Wih_b, bias, xg);

  lstm_scan_kernel<<<NBLK_, 256, 0, stream>>>(
      xg, Whh_b, h0, h1, cn, out, hn, counter);
}
